// MHSA_72997264163290
// MI455X (gfx1250) — matrix operand order repaired and checked by reading, not compiled, not run on hardware
//
#include <hip/hip_runtime.h>
#include <hip/hip_bf16.h>

typedef __attribute__((ext_vector_type(16))) _Float16 v16h;
typedef __attribute__((ext_vector_type(8)))  float    v8f;
typedef _Float16 f16;

#define CDIV(a,b) (((a)+(b)-1)/(b))

// Problem constants
#define CN   384
#define BN   4
#define NH   8
#define HD   48
#define HDP  64          // HD padded to 64 so K-dim = 2 WMMA chunks of 32
#define HW   4096        // 64*64 queries per image
#define NKEY 1024        // 32*32 keys per image
#define N1   16384       // BN*HW
#define N2   4096        // BN*NKEY
#define K4   1536        // C * 2*2 (im2col K for SR conv)
#define SCALE 0.14433756729740643f   // 48^-0.5

__device__ __forceinline__ v8f wmma16(v16h a, v16h b, v8f c) {
  // v_wmma_f32_16x16x32_f16 : D = A(16x32 f16) * B(32x16 f16) + C(16x16 f32)
  return __builtin_amdgcn_wmma_f32_16x16x32_f16(false, a, false, b, (short)0, c,
                                                false, false);
}

// Row reductions within 16-lane halves (wave32: rows of a 16x16 D tile live in
// lanes 0-15 (M=j) and lanes 16-31 (M=j+8); xor masks 1,2,4,8 stay in-half).
__device__ __forceinline__ float hmax16(float v) {
#pragma unroll
  for (int m = 1; m < 16; m <<= 1) v = fmaxf(v, __shfl_xor(v, m, 32));
  return v;
}
__device__ __forceinline__ float hsum16(float v) {
#pragma unroll
  for (int m = 1; m < 16; m <<= 1) v += __shfl_xor(v, m, 32);
  return v;
}

// A-fragment element->K mapping used here (16-bit A 16x32; contiguous, not the ISA table order, same as B):
// lane<16 holds K = 0..15 ; lane>=16 holds K = 16..31
__device__ __forceinline__ int a_k_of(int lane, int i) {
  return ((lane >= 16) ? 16 : 0) + i;
}
// B-fragment element->K mapping (16-bit B 32x16): lane<16: K=i, lane>=16: K=16+i
__device__ __forceinline__ int b_k_of(int lane, int i) {
  return ((lane >= 16) ? 16 : 0) + i;
}

// ---------------------------------------------------------------------------
// Pack kernels: build WMMA-fragment-ordered f16 blobs.
// A-pack layout: [mt][kt][lane(32)][16 f16]   (one v16h per lane per tile)
// B-pack layout: [kt][nt][lane(32)][16 f16]
// ---------------------------------------------------------------------------
__global__ void pack_A_f32(const float* __restrict__ W, f16* __restrict__ Ap,
                           int M, int K) {
  int ktn = K >> 5;
  int idx = blockIdx.x * blockDim.x + threadIdx.x;
  int total = (M >> 4) * ktn * 32;
  if (idx >= total) return;
  int lane = idx & 31, tile = idx >> 5;
  int kt = tile % ktn, mt = tile / ktn;
  int m = mt * 16 + (lane & 15);
  v16h out;
#pragma unroll
  for (int i = 0; i < 16; ++i) {
    int k = kt * 32 + a_k_of(lane, i);
    out[i] = (f16)W[(size_t)m * K + k];
  }
  ((v16h*)Ap)[idx] = out;
}

// B-pack of x viewed as [K=c][N=b*4096+hw]  (f32 NCHW source)
__global__ void pack_x_B(const float* __restrict__ x, f16* __restrict__ Bp) {
  int idx = blockIdx.x * blockDim.x + threadIdx.x;  // (kt*ntn+nt)*32+lane
  int total = (CN / 32) * (N1 / 16) * 32;
  if (idx >= total) return;
  int lane = idx & 31, tile = idx >> 5;
  int ntn = N1 / 16;
  int nt = tile % ntn, kt = tile / ntn;
  int n = nt * 16 + (lane & 15);
  int b = n >> 12, hw = n & 4095;
  v16h out;
#pragma unroll
  for (int i = 0; i < 16; ++i) {
    int c = kt * 32 + b_k_of(lane, i);
    out[i] = (f16)x[((size_t)(b * CN + c)) * 4096 + hw];
  }
  ((v16h*)Bp)[idx] = out;
}

// B-pack of im2col for the SR conv: K = c*4 + i2*2 + j, N = b*1024 + p
__global__ void pack_col_B(const float* __restrict__ x, f16* __restrict__ Bp) {
  int idx = blockIdx.x * blockDim.x + threadIdx.x;
  int total = (K4 / 32) * (N2 / 16) * 32;
  if (idx >= total) return;
  int lane = idx & 31, tile = idx >> 5;
  int ntn = N2 / 16;
  int nt = tile % ntn, kt = tile / ntn;
  int n = nt * 16 + (lane & 15);
  int b = n >> 10, p = n & 1023;
  int ph = p >> 5, pw = p & 31;
  v16h out;
#pragma unroll
  for (int i = 0; i < 16; ++i) {
    int kk = kt * 32 + b_k_of(lane, i);
    int c = kk >> 2, i2 = (kk >> 1) & 1, j = kk & 1;
    out[i] = (f16)x[((size_t)(b * CN + c)) * 4096 + (2 * ph + i2) * 64 +
                    (2 * pw + j)];
  }
  ((v16h*)Bp)[idx] = out;
}

// B-pack of a plain f16 row-major [K][N] matrix
__global__ void pack_h16_B(const f16* __restrict__ src, f16* __restrict__ Bp,
                           int K, int N) {
  int idx = blockIdx.x * blockDim.x + threadIdx.x;
  int total = (K >> 5) * (N >> 4) * 32;
  if (idx >= total) return;
  int lane = idx & 31, tile = idx >> 5;
  int ntn = N >> 4;
  int nt = tile % ntn, kt = tile / ntn;
  int n = nt * 16 + (lane & 15);
  v16h out;
#pragma unroll
  for (int i = 0; i < 16; ++i) {
    int k = kt * 32 + b_k_of(lane, i);
    out[i] = src[(size_t)k * N + n];
  }
  ((v16h*)Bp)[idx] = out;
}

// ---------------------------------------------------------------------------
// Generic WMMA GEMM: C[M][N] = A[M][K] * B[K][N] (+bias[m]).
// Register-blocked: one 16x64 output tile per wave (A-frag reused across the
// 4 contiguous B-frags of the supertile -> 4 back-to-back WMMAs per K-chunk).
// FINAL=true: f32 epilogue straight into NCHW d_out; else plain f16 [M][N].
// Requires N % 64 == 0.
// ---------------------------------------------------------------------------
template <bool FINAL>
__global__ void gemm_wmma(const f16* __restrict__ Ap, const f16* __restrict__ Bp,
                          f16* __restrict__ outH, float* __restrict__ outF,
                          const float* __restrict__ bias, int M, int N, int K) {
  int wave = blockIdx.x * (blockDim.x >> 5) + (threadIdx.x >> 5);
  int lane = threadIdx.x & 31;
  int ntn = N >> 4;    // 16-wide B tiles
  int nt4n = N >> 6;   // 64-wide supertiles
  int tiles = (M >> 4) * nt4n;
  if (wave >= tiles) return;
  int mt = wave / nt4n, nt0 = (wave % nt4n) * 4;
  int ktn = K >> 5;
  const v16h* Av = (const v16h*)Ap + ((size_t)mt * ktn) * 32 + lane;
  const v16h* Bv = (const v16h*)Bp + (size_t)nt0 * 32 + lane;
  v8f acc[4] = {{}, {}, {}, {}};
  for (int kt = 0; kt < ktn; ++kt) {
    v16h a = Av[(size_t)kt * 32];
    const v16h* bv = Bv + (size_t)kt * ntn * 32;
    acc[0] = wmma16(a, bv[0],  acc[0]);
    acc[1] = wmma16(a, bv[32], acc[1]);
    acc[2] = wmma16(a, bv[64], acc[2]);
    acc[3] = wmma16(a, bv[96], acc[3]);
  }
  int row0 = (lane < 16) ? 0 : 8;
#pragma unroll
  for (int t = 0; t < 4; ++t) {
    int col = (nt0 + t) * 16 + (lane & 15);
#pragma unroll
    for (int j = 0; j < 8; ++j) {
      int m = mt * 16 + row0 + j;
      float v = acc[t][j] + (bias ? bias[m] : 0.0f);
      if (FINAL) {
        int b = col >> 12, hw = col & 4095;  // col = b*4096 + hw
        outF[(((size_t)b * CN + m) << 12) + hw] = v;
      } else {
        outH[(size_t)m * N + col] = (f16)v;
      }
    }
  }
}

// ---------------------------------------------------------------------------
// q/k/v fragment builders (pad HD 48 -> 64 with zeros).
// qF: A-pack per (b,h): [b][h][qt(256)][kt(2)][lane][16]
// kF: B-pack per (b,h): [b][h][mt(64)][kt(2)][lane][16], K-dim = d; pos+scale fused
// vF: B-pack per (b,h): [b][h][mc(32)][nt(4)][lane][16], K-dim = keys
// ---------------------------------------------------------------------------
__global__ void make_q_frag(const f16* __restrict__ qh, f16* __restrict__ qF) {
  int idx = blockIdx.x * blockDim.x + threadIdx.x;
  if (idx >= BN * NH * 256 * 2 * 32) return;
  int lane = idx & 31, t = idx >> 5;
  int kt = t & 1;  t >>= 1;
  int qt = t & 255; t >>= 8;
  int h = t & 7;   int b = t >> 3;
  int q = qt * 16 + (lane & 15);
  v16h out;
#pragma unroll
  for (int i = 0; i < 16; ++i) {
    int d = kt * 32 + a_k_of(lane, i);
    out[i] = (d < HD) ? qh[(size_t)(h * HD + d) * N1 + b * HW + q] : (f16)0.0f;
  }
  ((v16h*)qF)[idx] = out;
}

__global__ void make_k_frag(const f16* __restrict__ kh,
                            const float* __restrict__ rel_h,
                            const float* __restrict__ rel_w,
                            f16* __restrict__ kF) {
  int idx = blockIdx.x * blockDim.x + threadIdx.x;
  if (idx >= BN * NH * 64 * 2 * 32) return;
  int lane = idx & 31, t = idx >> 5;
  int kt = t & 1; t >>= 1;
  int mt = t & 63; t >>= 6;
  int h = t & 7;  int b = t >> 3;
  int key = mt * 16 + (lane & 15);
  v16h out;
#pragma unroll
  for (int i = 0; i < 16; ++i) {
    int d = kt * 32 + b_k_of(lane, i);
    if (d < HD) {
      float kv = (float)kh[(size_t)(h * HD + d) * N2 + b * NKEY + key];
      // pos reshaped from (Ws,Hs): m = w*Hs + hs  (Hs = Ws = 32)
      float pos = rel_h[(size_t)(h * HD + d) * 32 + (key & 31)] +
                  rel_w[(size_t)(h * HD + d) * 32 + (key >> 5)];
      out[i] = (f16)((kv + pos) * SCALE);
    } else {
      out[i] = (f16)0.0f;
    }
  }
  ((v16h*)kF)[idx] = out;
}

__global__ void make_v_frag(const f16* __restrict__ vh, f16* __restrict__ vF) {
  int idx = blockIdx.x * blockDim.x + threadIdx.x;
  if (idx >= BN * NH * 32 * 4 * 32) return;
  int lane = idx & 31, t = idx >> 5;
  int nt = t & 3;  t >>= 2;
  int mc = t & 31; t >>= 5;
  int h = t & 7;   int b = t >> 3;
  int d = nt * 16 + (lane & 15);
  v16h out;
#pragma unroll
  for (int i = 0; i < 16; ++i) {
    int key = mc * 32 + b_k_of(lane, i);
    out[i] = (d < HD) ? vh[(size_t)(h * HD + d) * N2 + b * NKEY + key]
                      : (f16)0.0f;
  }
  ((v16h*)vF)[idx] = out;
}

// ---------------------------------------------------------------------------
// Flash attention: block = (b, h, 64-query slab), 4 waves x 16-query tiles.
// Per 32-key chunk: 2x2 WMMAs for S, online softmax, P via LDS reshuffle,
// 4 WMMAs for PV into fp32 accumulators.
// ---------------------------------------------------------------------------
__global__ void __launch_bounds__(128)
flash_attn(const f16* __restrict__ qF, const f16* __restrict__ kF,
           const f16* __restrict__ vF, f16* __restrict__ aoP) {
  __shared__ f16 plds[4][16][32];  // per-wave 16x32 P staging
  int wid = threadIdx.x >> 5, lane = threadIdx.x & 31;
  int blk = blockIdx.x;
  int qb = blk & 63;
  int h = (blk >> 6) & 7;
  int bb = blk >> 9;
  int qt = qb * 4 + wid;  // 16-query tile index in [0,256)

  const v16h* qFp =
      (const v16h*)qF + (((size_t)(bb * NH + h) * 256 + qt) * 2) * 32 + lane;
  v16h qa0 = qFp[0];
  v16h qa1 = qFp[32];
  const v16h* kFb = (const v16h*)kF + (size_t)(bb * NH + h) * 64 * 2 * 32;
  const v16h* vFb = (const v16h*)vF + (size_t)(bb * NH + h) * 32 * 4 * 32;

  v8f o0 = {}, o1 = {}, o2 = {}, o3 = {};
  float mrow[8], lrow[8];
#pragma unroll
  for (int j = 0; j < 8; ++j) { mrow[j] = -3.0e30f; lrow[j] = 0.0f; }

  for (int mc = 0; mc < 32; ++mc) {
    // S tiles for key tiles 2*mc and 2*mc+1 (each 16 keys, K-dim = 64 dims)
    v8f s0 = {}, s1 = {};
    {
      v16h b0 = kFb[((2 * mc) * 2 + 0) * 32 + lane];
      v16h b1 = kFb[((2 * mc) * 2 + 1) * 32 + lane];
      s0 = wmma16(qa0, b0, s0);
      s0 = wmma16(qa1, b1, s0);
      v16h b2 = kFb[((2 * mc + 1) * 2 + 0) * 32 + lane];
      v16h b3 = kFb[((2 * mc + 1) * 2 + 1) * 32 + lane];
      s1 = wmma16(qa0, b2, s1);
      s1 = wmma16(qa1, b3, s1);
    }
    // Online softmax, per D-tile row j (row M = j or j+8 depending on half)
#pragma unroll
    for (int j = 0; j < 8; ++j) {
      float mx = hmax16(fmaxf(s0[j], s1[j]));
      float nm = fmaxf(mrow[j], mx);
      float sc = __expf(mrow[j] - nm);
      mrow[j] = nm;
      float p0 = __expf(s0[j] - nm);
      float p1 = __expf(s1[j] - nm);
      s0[j] = p0; s1[j] = p1;
      lrow[j] = lrow[j] * sc + hsum16(p0 + p1);
      o0[j] *= sc; o1[j] *= sc; o2[j] *= sc; o3[j] *= sc;
    }
    // Reshape P (D-layout) -> A-layout via LDS
    int rbase = (lane < 16) ? 0 : 8;
    int ch = lane & 15;
#pragma unroll
    for (int j = 0; j < 8; ++j) {
      plds[wid][rbase + j][ch] = (f16)s0[j];
      plds[wid][rbase + j][16 + ch] = (f16)s1[j];
    }
    __syncthreads();
    v16h pa;
    int r = lane & 15;
#pragma unroll
    for (int i = 0; i < 16; ++i) pa[i] = plds[wid][r][a_k_of(lane, i)];
    __syncthreads();
    // PV: accumulate O (16 queries x 64 dims)
    o0 = wmma16(pa, vFb[(mc * 4 + 0) * 32 + lane], o0);
    o1 = wmma16(pa, vFb[(mc * 4 + 1) * 32 + lane], o1);
    o2 = wmma16(pa, vFb[(mc * 4 + 2) * 32 + lane], o2);
    o3 = wmma16(pa, vFb[(mc * 4 + 3) * 32 + lane], o3);
  }
  // Epilogue: O /= l, write plain f16 [c = h*48+d][b*4096 + q]; d>=48 padding dropped
  int rbase = (lane < 16) ? 0 : 8;
  int ch = lane & 15;
#pragma unroll
  for (int j = 0; j < 8; ++j) {
    float inv = 1.0f / lrow[j];
    int q = qt * 16 + rbase + j;
    size_t ncol = (size_t)bb * HW + q;
    aoP[(size_t)(h * HD + ch) * N1 + ncol] = (f16)(o0[j] * inv);
    aoP[(size_t)(h * HD + 16 + ch) * N1 + ncol] = (f16)(o1[j] * inv);
    aoP[(size_t)(h * HD + 32 + ch) * N1 + ncol] = (f16)(o2[j] * inv);
    // o3 holds d = 48..63 (zero padding) — discarded
  }
}

// ---------------------------------------------------------------------------
extern "C" void kernel_launch(void* const* d_in, const int* in_sizes, int n_in,
                              void* d_out, int out_size, void* d_ws,
                              size_t ws_size, hipStream_t stream) {
  const float* x     = (const float*)d_in[0];
  const float* Wq    = (const float*)d_in[1];
  const float* Wk    = (const float*)d_in[2];
  const float* Wv    = (const float*)d_in[3];
  const float* Wsr   = (const float*)d_in[4];  // (C,C,2,2) == [384][1536] row-major
  const float* bsr   = (const float*)d_in[5];
  const float* Wp    = (const float*)d_in[6];
  const float* bp    = (const float*)d_in[7];
  const float* rel_h = (const float*)d_in[8];
  const float* rel_w = (const float*)d_in[9];

  // workspace layout (f16 blobs); lifetime-based aliasing noted inline
  char* p = (char*)d_ws;
  const size_t SZ_WA   = (size_t)24 * 12 * 512 * 2;   // 288 KB
  const size_t SZ_WSRA = (size_t)24 * 48 * 512 * 2;   // 1.13 MB
  const size_t SZ_XB   = (size_t)12 * 1024 * 512 * 2; // 12 MB
  const size_t SZ_COLB = (size_t)48 * 256 * 512 * 2;  // 12 MB
  const size_t SZ_QH   = (size_t)CN * N1 * 2;         // 12 MB
  const size_t SZ_P4   = (size_t)CN * N2 * 2;         // 3 MB
  const size_t SZ_QF   = (size_t)BN * NH * 256 * 2 * 512 * 2;  // 16 MB
  const size_t SZ_KF   = (size_t)BN * NH * 64 * 2 * 512 * 2;   // 4 MB

  f16* WqA  = (f16*)p; p += SZ_WA;
  f16* WkA  = (f16*)p; p += SZ_WA;
  f16* WvA  = (f16*)p; p += SZ_WA;
  f16* WpA  = (f16*)p; p += SZ_WA;
  f16* WsrA = (f16*)p; p += SZ_WSRA;
  f16* xB   = (f16*)p; p += SZ_XB;    // dead after q GEMM -> reused as aoB
  f16* colB = (f16*)p; p += SZ_COLB;  // dead after SR GEMM -> reused as kF+vF
  f16* qh   = (f16*)p; p += SZ_QH;    // dead after q_frag -> reused as aoP
  f16* xsrP = (f16*)p; p += SZ_P4;
  f16* xsrB = (f16*)p; p += SZ_P4;
  f16* kh   = (f16*)p; p += SZ_P4;
  f16* vh   = (f16*)p; p += SZ_P4;
  f16* qF   = (f16*)p; p += SZ_QF;
  f16* kF   = colB;
  f16* vF   = colB + SZ_KF / sizeof(f16);
  f16* aoP  = qh;
  f16* aoB  = xB;

  // 1) pack weights + activations
  pack_A_f32<<<CDIV(24 * 12 * 32, 256), 256, 0, stream>>>(Wq, WqA, CN, CN);
  pack_A_f32<<<CDIV(24 * 12 * 32, 256), 256, 0, stream>>>(Wk, WkA, CN, CN);
  pack_A_f32<<<CDIV(24 * 12 * 32, 256), 256, 0, stream>>>(Wv, WvA, CN, CN);
  pack_A_f32<<<CDIV(24 * 12 * 32, 256), 256, 0, stream>>>(Wp, WpA, CN, CN);
  pack_A_f32<<<CDIV(24 * 48 * 32, 256), 256, 0, stream>>>(Wsr, WsrA, CN, K4);
  pack_x_B<<<CDIV(12 * 1024 * 32, 256), 256, 0, stream>>>(x, xB);
  pack_col_B<<<CDIV(48 * 256 * 32, 256), 256, 0, stream>>>(x, colB);

  // 2) q projection (M=384,N=16384,K=384): 24*256 supertiles, 8 waves/block
  gemm_wmma<false><<<CDIV(24 * 256, 8), 256, 0, stream>>>(
      WqA, xB, qh, nullptr, nullptr, CN, N1, CN);
  // 3) SR conv as GEMM (M=384,N=4096,K=1536) + bsr: 24*64 supertiles
  gemm_wmma<false><<<CDIV(24 * 64, 8), 256, 0, stream>>>(
      WsrA, colB, xsrP, nullptr, bsr, CN, N2, K4);
  // 4) repack xsr, project k and v
  pack_h16_B<<<CDIV(12 * 256 * 32, 256), 256, 0, stream>>>(xsrP, xsrB, CN, N2);
  gemm_wmma<false><<<CDIV(24 * 64, 8), 256, 0, stream>>>(
      WkA, xsrB, kh, nullptr, nullptr, CN, N2, CN);
  gemm_wmma<false><<<CDIV(24 * 64, 8), 256, 0, stream>>>(
      WvA, xsrB, vh, nullptr, nullptr, CN, N2, CN);
  // 5) build attention fragments (pos bias + scale fused into K)
  make_q_frag<<<CDIV(BN * NH * 256 * 2 * 32, 256), 256, 0, stream>>>(qh, qF);
  make_k_frag<<<CDIV(BN * NH * 64 * 2 * 32, 256), 256, 0, stream>>>(
      kh, rel_h, rel_w, kF);
  make_v_frag<<<CDIV(BN * NH * 32 * 4 * 32, 256), 256, 0, stream>>>(vh, vF);
  // 6) flash attention: 4*8*64 blocks of 4 waves
  flash_attn<<<BN * NH * 64, 128, 0, stream>>>(qF, kF, vF, aoP);
  // 7) output projection -> d_out (f32, NCHW) + bp
  pack_h16_B<<<CDIV(12 * 1024 * 32, 256), 256, 0, stream>>>(aoP, aoB, CN, N1);
  gemm_wmma<true><<<CDIV(24 * 256, 8), 256, 0, stream>>>(
      WpA, aoB, nullptr, (float*)d_out, bp, CN, N1, CN);
}
